// WeightedSumGraphRepresentation_49735721288008
// MI455X (gfx1250) — matrix operand order repaired and checked by reading, not compiled, not run on hardware
//
#include <hip/hip_runtime.h>
#include <hip/hip_bf16.h>

#define N_NODES  131072
#define G_SEGS   2048
#define D_IN     832
#define H_DIM    256
#define HEADS    8
#define REPR     416
#define SLOPE    0.01f

typedef __attribute__((ext_vector_type(16))) _Float16 v16h;
typedef __attribute__((ext_vector_type(8)))  float    v8f;
typedef __attribute__((ext_vector_type(4)))  unsigned v4u;

// ---------------------------------------------------------------------------
// Weight-conversion kernels (fp32 -> f16 packs, zero-padded to 64-col multiple)
// ---------------------------------------------------------------------------

// W1h[k][c] = c<256 ? sW1[k][c] : tW1[k][c-256];  shape [832][512]
__global__ __launch_bounds__(256) void concat_w1(const float* __restrict__ sW1,
                                                 const float* __restrict__ tW1,
                                                 _Float16* __restrict__ dst) {
    int i = blockIdx.x * 256 + threadIdx.x;
    if (i >= D_IN * 512) return;
    int k = i >> 9;
    int c = i & 511;
    float v = (c < H_DIM) ? sW1[k * H_DIM + c] : tW1[k * H_DIM + (c - H_DIM)];
    dst[i] = (_Float16)v;
}

__global__ __launch_bounds__(256) void concat_b1(const float* __restrict__ sb1,
                                                 const float* __restrict__ tb1,
                                                 float* __restrict__ dst) {
    int i = threadIdx.x + blockIdx.x * 256;
    if (i >= 512) return;
    dst[i] = (i < H_DIM) ? sb1[i] : tb1[i - H_DIM];
}

__global__ __launch_bounds__(256) void cvt_f32_f16(const float* __restrict__ src,
                                                   _Float16* __restrict__ dst, int n) {
    int i = blockIdx.x * 256 + threadIdx.x;
    if (i < n) dst[i] = (_Float16)src[i];
}

// generic pad: src [K x ncol] fp32 -> dst [K x ncolPad] f16, zeros beyond ncol
__global__ __launch_bounds__(256) void pad_cvt(const float* __restrict__ src,
                                               _Float16* __restrict__ dst,
                                               int ncol, int ncolPad, int total) {
    int i = blockIdx.x * 256 + threadIdx.x;
    if (i >= total) return;
    int k = i / ncolPad, c = i - k * ncolPad;
    dst[i] = (c < ncol) ? (_Float16)src[k * ncol + c] : (_Float16)0.0f;
}

// ---------------------------------------------------------------------------
// Tiled WMMA GEMM:  C[M x Nstore] = act(A[M x K] @ B[K x ldbP] + bias)
//   block = 256 threads (8 waves); tile = 128 rows x 64 cols; BK = 32
//   Double-buffered LDS; B staged fragment-ready (2x ds_load_b128 / fragment).
//   ASYNC=true (f16 A): A tile staged with global_load_async_to_lds_b128.
//   NOTE: As must be the FIRST shared object so its LDS byte offset == index.
// ---------------------------------------------------------------------------
template <typename AT, typename OT, bool RELU, bool ASYNC>
__global__ __launch_bounds__(256) void gemm_wmma(const AT* __restrict__ A, int lda,
                                                 const _Float16* __restrict__ B, int ldbP,
                                                 const float* __restrict__ bias,
                                                 OT* __restrict__ C, int ldc, int Nstore,
                                                 int K) {
    __shared__ __align__(16) _Float16 As[2][128 * 32];    // 16 KB (offset 0 in LDS)
    __shared__ __align__(16) unsigned Bsw[2][4 * 32 * 8]; //  8 KB fragment-ready

    const int tid   = threadIdx.x;
    const int lane  = tid & 31;
    const int wave  = tid >> 5;
    const int g     = lane >> 4;
    const int r     = lane & 15;

    const int rowBlk = blockIdx.y * 128;
    const int colBlk = blockIdx.x * 64;

    v8f acc[4];
    #pragma unroll
    for (int ct = 0; ct < 4; ++ct) acc[ct] = (v8f)0.0f;

    // A staging map: thread -> (row, 16-wide k chunk)
    const int a_row = tid >> 1;
    const int a_kp  = (tid & 1) * 16;
    const AT* aSrcBase = A + (size_t)(rowBlk + a_row) * lda + a_kp;

    // B staging map: thread -> (column, 8-row k chunk); branch-free (B padded)
    const int c_lin = tid & 63;
    const int p     = tid >> 6;
    const int cgOff = colBlk + c_lin;           // column into padded B
    const int ctS   = c_lin >> 4;
    const unsigned short* Bu = (const unsigned short*)B;

    auto stage = [&](int buf, int k0) {
        // ---- A tile ----
        if (ASYNC) {
            // pure 32B/lane copy: async global->LDS, no VGPR round trip
            unsigned ldsA = (unsigned)(buf * (128 * 32 * 2) + (a_row * 32 + a_kp) * 2);
            const AT* gsrc = aSrcBase + k0;
            asm volatile("global_load_async_to_lds_b128 %0, %1, off"
                         :: "v"(ldsA), "v"(gsrc) : "memory");
            asm volatile("global_load_async_to_lds_b128 %0, %1, off offset:16"
                         :: "v"(ldsA), "v"(gsrc) : "memory");
        } else {
            const AT* src = aSrcBase + k0;
            if (k0 + 64 < K) __builtin_prefetch(src + 64);
            _Float16* dst = &As[buf][a_row * 32 + a_kp];
            #pragma unroll
            for (int i = 0; i < 16; ++i) dst[i] = (_Float16)src[i];
        }
        // ---- B tile: pack k/k+1 half pairs into fragment-ready dwords ----
        #pragma unroll
        for (int q2 = 0; q2 < 4; ++q2) {
            const int kk = p * 8 + 2 * q2;
            unsigned lo = Bu[(k0 + kk) * ldbP + cgOff];
            unsigned hi = Bu[(k0 + kk + 1) * ldbP + cgOff];
            const int gS  = kk >> 4;
            const int vS  = (kk & 15) >> 1;
            const int lnS = gS * 16 + (c_lin & 15);
            Bsw[buf][(ctS * 32 + lnS) * 8 + vS] = lo | (hi << 16);
        }
    };

    stage(0, 0);
    if (ASYNC) asm volatile("s_wait_asynccnt 0x0" ::: "memory");
    __syncthreads();

    int buf = 0;
    const int rb = (wave * 16 + r) * 16;   // A row base in dwords

    for (int k0 = 0; k0 < K; k0 += 32) {
        if (k0 + 32 < K) stage(buf ^ 1, k0 + 32);

        // ---- pre-load all fragments (DS loads clause together) ----
        const unsigned* Asu = (const unsigned*)As[buf];
        union F { v16h v; v4u q[2]; };
        F af;
        af.q[0] = *(const v4u*)&Asu[rb + g * 8];
        af.q[1] = *(const v4u*)&Asu[rb + g * 8 + 4];
        F bf[4];
        #pragma unroll
        for (int ct = 0; ct < 4; ++ct) {
            const v4u* bp = (const v4u*)&Bsw[buf][(ct * 32 + lane) * 8];
            bf[ct].q[0] = bp[0];
            bf[ct].q[1] = bp[1];
        }
        // ---- back-to-back WMMAs ----
        #pragma unroll
        for (int ct = 0; ct < 4; ++ct)
            acc[ct] = __builtin_amdgcn_wmma_f32_16x16x32_f16(
                false, af.v, false, bf[ct].v, (short)0, acc[ct], false, false);

        if (ASYNC) asm volatile("s_wait_asynccnt 0x0" ::: "memory");
        __syncthreads();
        buf ^= 1;
    }

    // ---- epilogue: bias + LeakyReLU, guarded store ----
    #pragma unroll
    for (int ct = 0; ct < 4; ++ct) {
        const int colg = colBlk + ct * 16 + r;
        if (colg >= Nstore) continue;
        const float bv = bias[colg];
        #pragma unroll
        for (int rr = 0; rr < 8; ++rr) {
            const int rowg = rowBlk + wave * 16 + g * 8 + rr;
            float f = acc[ct][rr] + bv;
            if (RELU) f = (f > 0.0f) ? f : SLOPE * f;
            C[(size_t)rowg * ldc + colg] = (OT)f;
        }
    }
}

// ---------------------------------------------------------------------------
// Segment starts: batch is sorted; seg_start[g] = lower_bound(batch, g)
// ---------------------------------------------------------------------------
__global__ __launch_bounds__(256) void seg_starts(const int* __restrict__ batch,
                                                  int* __restrict__ segst) {
    int gIdx = blockIdx.x * 256 + threadIdx.x;
    if (gIdx > G_SEGS) return;
    if (gIdx == G_SEGS) { segst[gIdx] = N_NODES; return; }
    int lo = 0, hi = N_NODES;
    while (lo < hi) {
        int mid = (lo + hi) >> 1;
        if (batch[mid] < gIdx) lo = mid + 1; else hi = mid;
    }
    segst[gIdx] = lo;
}

// ---------------------------------------------------------------------------
// Per-graph softmax (per head) + weighted segment-sum -> out[G, 416]
// ---------------------------------------------------------------------------
__global__ __launch_bounds__(256) void pool_graphs(const float* __restrict__ scores,
                                                   const _Float16* __restrict__ nodeR,
                                                   const int* __restrict__ segst,
                                                   float* __restrict__ out) {
    __shared__ float red[256];
    __shared__ float smax[HEADS];
    __shared__ float srcp[HEADS];

    const int gid = blockIdx.x;
    const int s = segst[gid], e = segst[gid + 1];
    const int tid = threadIdx.x;
    const int head = tid & 7;
    const int j = tid >> 3;

    float m = -1e30f;
    for (int n = s + j; n < e; n += 32) m = fmaxf(m, scores[(size_t)n * HEADS + head]);
    red[tid] = m;
    __syncthreads();
    for (int st = 16; st > 0; st >>= 1) {
        if (j < st) red[tid] = fmaxf(red[tid], red[tid + st * 8]);
        __syncthreads();
    }
    if (tid < HEADS) smax[tid] = red[tid];
    __syncthreads();

    float sum = 0.0f;
    const float mh = smax[head];
    for (int n = s + j; n < e; n += 32) sum += __expf(scores[(size_t)n * HEADS + head] - mh);
    red[tid] = sum;
    __syncthreads();
    for (int st = 16; st > 0; st >>= 1) {
        if (j < st) red[tid] += red[tid + st * 8];
        __syncthreads();
    }
    if (tid < HEADS) srcp[tid] = (e > s) ? (1.0f / red[tid]) : 0.0f;
    __syncthreads();

    for (int c = tid; c < REPR; c += 256) {
        const int h = c / (REPR / HEADS);
        const float mx = smax[h], rc = srcp[h];
        float accv = 0.0f;
        for (int n = s; n < e; ++n) {
            float w = __expf(scores[(size_t)n * HEADS + h] - mx) * rc;
            accv += w * (float)nodeR[(size_t)n * REPR + c];
        }
        out[(size_t)gid * REPR + c] = accv;
    }
}

// ---------------------------------------------------------------------------
// Host-side launcher
// ---------------------------------------------------------------------------
extern "C" void kernel_launch(void* const* d_in, const int* in_sizes, int n_in,
                              void* d_out, int out_size, void* d_ws, size_t ws_size,
                              hipStream_t stream) {
    const float* x    = (const float*)d_in[0];
    const int*   batch= (const int*)  d_in[1];
    const float* sW1  = (const float*)d_in[2];
    const float* sb1  = (const float*)d_in[3];
    const float* sW2  = (const float*)d_in[4];
    const float* sb2  = (const float*)d_in[5];
    const float* sW3  = (const float*)d_in[6];
    const float* sb3  = (const float*)d_in[7];
    const float* tW1  = (const float*)d_in[8];
    const float* tb1  = (const float*)d_in[9];
    const float* tW2  = (const float*)d_in[10];
    const float* tb2  = (const float*)d_in[11];
    const float* tW3  = (const float*)d_in[12];
    const float* tb3  = (const float*)d_in[13];
    float* out = (float*)d_out;

    char* base = (char*)d_ws;
    size_t off = 0;
    auto carve = [&](size_t bytes) -> char* {
        char* ptr = base + off;
        off = (off + bytes + 255) & ~(size_t)255;
        return ptr;
    };
    _Float16* W1h  = (_Float16*)carve((size_t)D_IN * 512 * 2);
    float*    b1   = (float*)   carve(512 * 4);
    _Float16* W2sh = (_Float16*)carve((size_t)H_DIM * H_DIM * 2);
    _Float16* W2th = (_Float16*)carve((size_t)H_DIM * H_DIM * 2);
    _Float16* W3sh = (_Float16*)carve((size_t)H_DIM * 64 * 2);    // padded 8 -> 64
    _Float16* W3th = (_Float16*)carve((size_t)H_DIM * 448 * 2);   // padded 416 -> 448
    _Float16* h1   = (_Float16*)carve((size_t)N_NODES * 512 * 2);
    _Float16* h2s  = (_Float16*)carve((size_t)N_NODES * H_DIM * 2);
    _Float16* h2t  = (_Float16*)carve((size_t)N_NODES * H_DIM * 2);
    float*    sc   = (float*)   carve((size_t)N_NODES * HEADS * 4);
    _Float16* nR   = (_Float16*)carve((size_t)N_NODES * REPR * 2);
    int*      segst= (int*)     carve((G_SEGS + 1) * 4);
    (void)ws_size;

    // 1) weight conversion (zero-padded to 64-col multiples where needed)
    concat_w1<<<(D_IN * 512 + 255) / 256, 256, 0, stream>>>(sW1, tW1, W1h);
    concat_b1<<<2, 256, 0, stream>>>(sb1, tb1, b1);
    cvt_f32_f16<<<(H_DIM * H_DIM + 255) / 256, 256, 0, stream>>>(sW2, W2sh, H_DIM * H_DIM);
    cvt_f32_f16<<<(H_DIM * H_DIM + 255) / 256, 256, 0, stream>>>(tW2, W2th, H_DIM * H_DIM);
    pad_cvt<<<(H_DIM * 64 + 255) / 256, 256, 0, stream>>>(sW3, W3sh, HEADS, 64, H_DIM * 64);
    pad_cvt<<<(H_DIM * 448 + 255) / 256, 256, 0, stream>>>(tW3, W3th, REPR, 448, H_DIM * 448);

    const int MB = N_NODES / 128;   // 1024 row blocks

    // 2) L1: h1 = lrelu(x @ W1h + b1), [N,512] f16  (x streamed once, f32 A path)
    gemm_wmma<float, _Float16, true, false><<<dim3(512 / 64, MB), 256, 0, stream>>>(
        x, D_IN, W1h, 512, b1, h1, 512, 512, D_IN);

    // 3) L2: h2s / h2t = lrelu(h1[:, half] @ W2 + b2), [N,256] f16 (async A path)
    gemm_wmma<_Float16, _Float16, true, true><<<dim3(H_DIM / 64, MB), 256, 0, stream>>>(
        h1, 512, W2sh, H_DIM, sb2, h2s, H_DIM, H_DIM, H_DIM);
    gemm_wmma<_Float16, _Float16, true, true><<<dim3(H_DIM / 64, MB), 256, 0, stream>>>(
        h1 + H_DIM, 512, W2th, H_DIM, tb2, h2t, H_DIM, H_DIM, H_DIM);

    // 4) L3s: scores = h2s @ sW3 + sb3, [N,8] f32 (no activation)
    gemm_wmma<_Float16, float, false, true><<<dim3(1, MB), 256, 0, stream>>>(
        h2s, H_DIM, W3sh, 64, sb3, sc, HEADS, HEADS, H_DIM);

    //    L3t: nodeR = lrelu(h2t @ tW3 + tb3), [N,416] f16
    gemm_wmma<_Float16, _Float16, true, true><<<dim3(448 / 64, MB), 256, 0, stream>>>(
        h2t, H_DIM, W3th, 448, tb3, nR, REPR, REPR, H_DIM);

    // 5) segment boundaries (batch sorted)
    seg_starts<<<(G_SEGS + 1 + 255) / 256, 256, 0, stream>>>(batch, segst);

    // 6) per-graph softmax + weighted segment sum -> out [G, 416]
    pool_graphs<<<G_SEGS, 256, 0, stream>>>(sc, nR, segst, out);
}
